// Cross_transformer_15418932592904
// MI455X (gfx1250) — matrix operand order repaired and checked by reading, not compiled, not run on hardware
//
#include <hip/hip_runtime.h>

typedef __attribute__((ext_vector_type(16))) __bf16 v16bf;
typedef __attribute__((ext_vector_type(8)))  float  v8f;

#define NPIX 36864   // 192*192
#define CCH  48
#define BB   8

// ---------------------------------------------------------------------------
// WMMA helpers (CDNA5: V_WMMA_F32_16X16X32_BF16, wave32)
// ---------------------------------------------------------------------------
__device__ __forceinline__ v8f wmma_bf16(v16bf a, v16bf b, v8f c) {
  return __builtin_amdgcn_wmma_f32_16x16x32_bf16(
      /*neg_a=*/false, a, /*neg_b=*/false, b,
      /*c_mod=*/(short)0, c, /*reuse_a=*/false, /*reuse_b=*/false);
}

// A-matrix 16x32 bf16 layout (ISA 7.12.2): lane m=l%16; pair p holds
//   k = (p<4 ? 2p : 16+2(p-4)) + (lane>=16 ? 8 : 0), k and k+1.
__device__ __forceinline__ int frag_ka(int p, int lane) {
  int k = (p < 4) ? (p << 1) : (16 + ((p - 4) << 1));
  return k + (((lane >> 4) & 1) << 3);
}
// B-matrix 32x16 bf16 layout: lane n=l%16; same k order as A:
// pair p -> k = (p<4 ? 2p : 16+2(p-4)) + (lane>=16 ? 8 : 0).
__device__ __forceinline__ int frag_kb(int p, int lane) {
  int k = (p < 4) ? (p << 1) : (16 + ((p - 4) << 1));
  return k + (((lane >> 4) & 1) << 3);
}

// Load a 16x32 A fragment from row-major [row][k] storage (LDS or global).
__device__ __forceinline__ v16bf load_frag_a(const __bf16* base, int stride, int lane) {
  v16bf f;
  int r = lane & 15;
#pragma unroll
  for (int p = 0; p < 8; ++p) {
    int k = frag_ka(p, lane);
    f[2 * p]     = base[r * stride + k];
    f[2 * p + 1] = base[r * stride + k + 1];
  }
  return f;
}
// Load a 32x16 B fragment from col-major-by-n storage: element (k, n) at
// base[n*stride + k]  (n = output column, k = contraction index).
__device__ __forceinline__ v16bf load_frag_b(const __bf16* base, int stride, int lane) {
  v16bf f;
  int n = lane & 15;
#pragma unroll
  for (int p = 0; p < 8; ++p) {
    int k = frag_kb(p, lane);
    f[2 * p]     = base[n * stride + k];
    f[2 * p + 1] = base[n * stride + k + 1];
  }
  return f;
}

// ---------------------------------------------------------------------------
// Kernel 1: q/K projections.  out[s,b] = W_s (48x48) * x_s[b] (48xN), bf16 out.
// grid (N/128, 4*B), block 256 (8 waves; wave = n-tile, 3 m-tiles each).
// ---------------------------------------------------------------------------
__global__ __launch_bounds__(256) void qk_gemm_kernel(
    const float* __restrict__ x0, const float* __restrict__ x1,
    const float* __restrict__ x2, const float* __restrict__ x3,
    const float* __restrict__ W0, const float* __restrict__ W1,
    const float* __restrict__ W2, const float* __restrict__ W3,
    __bf16* __restrict__ qk) {
  int sb = blockIdx.y;
  int s = sb >> 3, b = sb & 7;
  const float* xs[4] = {x0, x1, x2, x3};
  const float* Ws[4] = {W0, W1, W2, W3};
  const float* x = xs[s] + (size_t)b * CCH * NPIX;
  const float* W = Ws[s];
  __bf16* out = qk + (size_t)sb * CCH * NPIX;

  __shared__ __bf16 lA[CCH * 64];   // [m][k], k padded 48->64 with zeros
  __shared__ __bf16 lB[128 * 64];   // [n][k]
  int tid = threadIdx.x;
  size_t n0 = (size_t)blockIdx.x * 128;

  for (int e = tid; e < CCH * 64; e += 256) {
    int m = e >> 6, k = e & 63;
    lA[e] = (k < CCH) ? (__bf16)W[m * CCH + k] : (__bf16)0.0f;
  }
  for (int e = tid; e < CCH * 128; e += 256) {
    int k = e >> 7, n = e & 127;
    lB[n * 64 + k] = (__bf16)x[(size_t)k * NPIX + n0 + n];
  }
  for (int e = tid; e < 16 * 128; e += 256) {
    int k = 48 + (e >> 7), n = e & 127;
    lB[n * 64 + k] = (__bf16)0.0f;
  }
  __syncthreads();

  int wave = tid >> 5, lane = tid & 31;
  int ni = wave;
  int hi = ((lane >> 4) & 1) << 3;
#pragma unroll
  for (int mi = 0; mi < 3; ++mi) {
    v8f acc = {};
#pragma unroll
    for (int kc = 0; kc < 2; ++kc) {
      v16bf a = load_frag_a(&lA[mi * 16 * 64 + kc * 32], 64, lane);
      v16bf bf = load_frag_b(&lB[ni * 16 * 64 + kc * 32], 64, lane);
      acc = wmma_bf16(a, bf, acc);
    }
#pragma unroll
    for (int v = 0; v < 8; ++v) {
      int row = mi * 16 + v + hi;
      out[(size_t)row * NPIX + n0 + ni * 16 + (lane & 15)] = (__bf16)acc[v];
    }
  }
}

// ---------------------------------------------------------------------------
// Kernel 2: energy[s,b,c,d] = sum_n q[b,c,n] * K[s,b,d,n].
// grid (32 chunks, 4*B), block 288 (9 waves = 3x3 output tiles).
// A/B fragments straight from global (contraction dim is contiguous).
// Partial tiles accumulated with global f32 atomics (buffer pre-zeroed).
// ---------------------------------------------------------------------------
__global__ __launch_bounds__(288) void energy_kernel(
    const __bf16* __restrict__ qk, float* __restrict__ energy) {
  int sb = blockIdx.y;
  int b = sb & 7;
  const __bf16* q  = qk + (size_t)b * CCH * NPIX;          // slot s=0 holds q
  const __bf16* Ks = qk + (size_t)sb * CCH * NPIX;
  int wave = threadIdx.x >> 5, lane = threadIdx.x & 31;
  int mi = wave / 3, di = wave % 3;
  v8f acc = {};
  int chunk0 = blockIdx.x * 36;
#pragma unroll 4
  for (int c = 0; c < 36; ++c) {
    size_t nb = (size_t)(chunk0 + c) * 32;
    v16bf a  = load_frag_a(q  + (size_t)(mi * 16) * NPIX + nb, NPIX, lane);
    v16bf bf = load_frag_b(Ks + (size_t)(di * 16) * NPIX + nb, NPIX, lane);
    acc = wmma_bf16(a, bf, acc);
  }
  float* e = energy + (size_t)sb * CCH * CCH;
  int hi = ((lane >> 4) & 1) << 3;
#pragma unroll
  for (int v = 0; v < 8; ++v) {
    int row = mi * 16 + v + hi;
    int col = di * 16 + (lane & 15);
    atomicAdd(&e[row * CCH + col], acc[v]);
  }
}

// ---------------------------------------------------------------------------
// Kernel 3: reversed-logit softmax.  attn = softmax(rowmax - e) over d.
// Stable form: exp(rowmin - e) / sum.  Output bf16, K-dim padded to 64.
// ---------------------------------------------------------------------------
__global__ void softmax_kernel(const float* __restrict__ energy,
                               __bf16* __restrict__ attn) {
  int sb = blockIdx.x;
  const float* e = energy + (size_t)sb * CCH * CCH;
  __bf16* a = attn + (size_t)sb * CCH * 64;
  int r = threadIdx.x;
  if (r < CCH) {
    float vmin = 1e30f;
    for (int d = 0; d < CCH; ++d) vmin = fminf(vmin, e[r * CCH + d]);
    float sum = 0.0f;
    for (int d = 0; d < CCH; ++d) sum += __expf(vmin - e[r * CCH + d]);
    float inv = 1.0f / sum;
    for (int d = 0; d < 64; ++d)
      a[r * 64 + d] =
          (d < CCH) ? (__bf16)(__expf(vmin - e[r * CCH + d]) * inv) : (__bf16)0.0f;
  }
}

// ---------------------------------------------------------------------------
// Kernel 4: out4[s,b] = attn[s,b] (48x48) * K[s,b] (48xN) -> cat[b, s*48.., n]
// ---------------------------------------------------------------------------
__global__ __launch_bounds__(256) void out4_gemm_kernel(
    const __bf16* __restrict__ qk, const __bf16* __restrict__ attn,
    __bf16* __restrict__ cat) {
  int sb = blockIdx.y;
  int s = sb >> 3, b = sb & 7;
  const __bf16* Ks = qk + (size_t)sb * CCH * NPIX;
  const __bf16* A  = attn + (size_t)sb * CCH * 64;  // already zero-padded K
  __bf16* out = cat + ((size_t)b * 4 * CCH + (size_t)s * CCH) * NPIX;

  __shared__ __bf16 lB[128 * 64];
  int tid = threadIdx.x;
  size_t n0 = (size_t)blockIdx.x * 128;
  for (int e = tid; e < CCH * 128; e += 256) {
    int k = e >> 7, n = e & 127;
    lB[n * 64 + k] = Ks[(size_t)k * NPIX + n0 + n];
  }
  for (int e = tid; e < 16 * 128; e += 256) {
    int k = 48 + (e >> 7), n = e & 127;
    lB[n * 64 + k] = (__bf16)0.0f;
  }
  __syncthreads();

  int wave = tid >> 5, lane = tid & 31;
  int ni = wave;
  int hi = ((lane >> 4) & 1) << 3;
#pragma unroll
  for (int mi = 0; mi < 3; ++mi) {
    v8f acc = {};
#pragma unroll
    for (int kc = 0; kc < 2; ++kc) {
      v16bf a  = load_frag_a(A + mi * 16 * 64 + kc * 32, 64, lane);
      v16bf bf = load_frag_b(&lB[ni * 16 * 64 + kc * 32], 64, lane);
      acc = wmma_bf16(a, bf, acc);
    }
#pragma unroll
    for (int v = 0; v < 8; ++v) {
      int row = mi * 16 + v + hi;
      out[(size_t)row * NPIX + n0 + ni * 16 + (lane & 15)] = (__bf16)acc[v];
    }
  }
}

// ---------------------------------------------------------------------------
// Kernel 5: fuse 1x1 conv: pre[b] = fuse_w (48x192) * cat[b] (192xN) + bias,
// plus per-channel sum/sumsq for BatchNorm (training stats).
// grid (N/64, B), block 384 (12 waves = 3 m-tiles x 4 n-tiles).
// ---------------------------------------------------------------------------
__global__ __launch_bounds__(384) void fuse_conv_kernel(
    const __bf16* __restrict__ cat, const float* __restrict__ fw,
    const float* __restrict__ fbias, float* __restrict__ pre,
    float* __restrict__ gsum, float* __restrict__ gssq) {
  int b = blockIdx.y;
  size_t n0 = (size_t)blockIdx.x * 64;
  const __bf16* xb = cat + (size_t)b * 4 * CCH * NPIX;

  __shared__ __bf16 lA[CCH * 192];   // [m][k]
  __shared__ __bf16 lB[64 * 192];    // [n][k]
  __shared__ float sSum[CCH], sSsq[CCH];
  int tid = threadIdx.x;
  if (tid < CCH) { sSum[tid] = 0.0f; sSsq[tid] = 0.0f; }
  for (int e = tid; e < CCH * 192; e += 384) lA[e] = (__bf16)fw[e];
  for (int e = tid; e < 192 * 64; e += 384) {
    int k = e >> 6, n = e & 63;
    lB[n * 192 + k] = xb[(size_t)k * NPIX + n0 + n];
  }
  __syncthreads();

  int wave = tid >> 5, lane = tid & 31;
  int mi = wave % 3, ni = wave / 3;   // ni 0..3
  v8f acc = {};
#pragma unroll
  for (int kc = 0; kc < 6; ++kc) {
    v16bf a  = load_frag_a(&lA[mi * 16 * 192 + kc * 32], 192, lane);
    v16bf bf = load_frag_b(&lB[ni * 16 * 192 + kc * 32], 192, lane);
    acc = wmma_bf16(a, bf, acc);
  }
  int hi = ((lane >> 4) & 1) << 3;
#pragma unroll
  for (int v = 0; v < 8; ++v) {
    int row = mi * 16 + v + hi;
    float y = acc[v] + fbias[row];
    pre[((size_t)b * CCH + row) * NPIX + n0 + ni * 16 + (lane & 15)] = y;
    atomicAdd(&sSum[row], y);
    atomicAdd(&sSsq[row], y * y);
  }
  __syncthreads();
  if (tid < CCH) {
    atomicAdd(&gsum[tid], sSum[tid]);
    atomicAdd(&gssq[tid], sSsq[tid]);
  }
}

// ---------------------------------------------------------------------------
// BN stats finalize: mean / inv-std per channel.
// ---------------------------------------------------------------------------
__global__ void stats_kernel(const float* __restrict__ sum,
                             const float* __restrict__ ssq,
                             float* __restrict__ mean,
                             float* __restrict__ istd, float inv_count) {
  int c = threadIdx.x;
  if (c < CCH) {
    float m = sum[c] * inv_count;
    float v = ssq[c] * inv_count - m * m;
    mean[c] = m;
    istd[c] = rsqrtf(v + 1e-5f);
  }
}

// ---------------------------------------------------------------------------
// Kernel 6: mid = gamma_cam * relu(BN(fuse_pre)) + input   (bf16 out)
// ---------------------------------------------------------------------------
__global__ __launch_bounds__(256) void bn_mid_kernel(
    const float* __restrict__ pre, const float* __restrict__ inp,
    const float* __restrict__ mean, const float* __restrict__ istd,
    const float* __restrict__ gam, const float* __restrict__ bet,
    const float* __restrict__ gcam, __bf16* __restrict__ mid) {
  size_t i = (size_t)blockIdx.x * 256 + threadIdx.x;
  int c = (int)((i / NPIX) % CCH);
  float y = pre[i];
  float a = gam[c] * (y - mean[c]) * istd[c] + bet[c];
  a = fmaxf(a, 0.0f);
  mid[i] = (__bf16)(gcam[0] * a + inp[i]);
}

// ---------------------------------------------------------------------------
// Kernel 7: 3x3 conv (pad 1) as implicit GEMM, 9 taps x 2 K-chunks = 18 WMMA
// per 16x16 tile.  Input halo tile staged in LDS as [ix][ky][c] so B-fragment
// pairs are contiguous.  Fused bias + BN stats.
// grid (W/64, H, B), block 384 (12 waves = 3 m-tiles x 4 n-tiles).
// ---------------------------------------------------------------------------
__global__ __launch_bounds__(384) void out_conv_kernel(
    const __bf16* __restrict__ mid, const float* __restrict__ w,
    const float* __restrict__ bias, float* __restrict__ pre,
    float* __restrict__ gsum, float* __restrict__ gssq) {
  int b = blockIdx.z, y = blockIdx.y;
  int x0 = blockIdx.x * 64;
  __shared__ __bf16 lIn[66 * 3 * 64];   // [ix][ky][c], c padded 48->64
  __shared__ __bf16 lA[CCH * 64];       // weights for one tap, [m][k]
  __shared__ float sSum[CCH], sSsq[CCH];
  int tid = threadIdx.x;
  if (tid < CCH) { sSum[tid] = 0.0f; sSsq[tid] = 0.0f; }
  const __bf16* mb = mid + (size_t)b * CCH * NPIX;

  for (int e = tid; e < 66 * 3 * 64; e += 384) {
    int c = e & 63;
    int t = e >> 6;
    int ky = t % 3, ix = t / 3;
    int inx = x0 - 1 + ix, iny = y - 1 + ky;
    __bf16 v = (__bf16)0.0f;
    if (c < CCH && inx >= 0 && inx < 192 && iny >= 0 && iny < 192)
      v = mb[(size_t)c * NPIX + (size_t)iny * 192 + inx];
    lIn[(ix * 3 + ky) * 64 + c] = v;
  }

  int wave = tid >> 5, lane = tid & 31;
  int mi = wave % 3, ni = wave / 3;   // ni 0..3
  v8f acc = {};
  for (int ky = 0; ky < 3; ++ky) {
    for (int kx = 0; kx < 3; ++kx) {
      __syncthreads();   // lIn ready (first tap) / lA of prev tap consumed
      for (int e = tid; e < CCH * 64; e += 384) {
        int m = e >> 6, k = e & 63;
        lA[e] = (k < CCH) ? (__bf16)w[((m * CCH + k) * 3 + ky) * 3 + kx]
                          : (__bf16)0.0f;
      }
      __syncthreads();
#pragma unroll
      for (int kc = 0; kc < 2; ++kc) {
        v16bf a  = load_frag_a(&lA[mi * 16 * 64 + kc * 32], 64, lane);
        v16bf bf = load_frag_b(&lIn[(ni * 16 + kx) * 192 + ky * 64 + kc * 32],
                               192, lane);
        acc = wmma_bf16(a, bf, acc);
      }
    }
  }
  int hi = ((lane >> 4) & 1) << 3;
#pragma unroll
  for (int v = 0; v < 8; ++v) {
    int row = mi * 16 + v + hi;
    float yv = acc[v] + bias[row];
    pre[((size_t)b * CCH + row) * NPIX + (size_t)y * 192 + x0 + ni * 16 +
        (lane & 15)] = yv;
    atomicAdd(&sSum[row], yv);
    atomicAdd(&sSsq[row], yv * yv);
  }
  __syncthreads();
  if (tid < CCH) {
    atomicAdd(&gsum[tid], sSum[tid]);
    atomicAdd(&gssq[tid], sSsq[tid]);
  }
}

// ---------------------------------------------------------------------------
// Kernel 8: final out = relu(BN(out_pre))  (fp32 out)
// ---------------------------------------------------------------------------
__global__ __launch_bounds__(256) void bn_out_kernel(
    const float* __restrict__ pre, const float* __restrict__ mean,
    const float* __restrict__ istd, const float* __restrict__ gam,
    const float* __restrict__ bet, float* __restrict__ out) {
  size_t i = (size_t)blockIdx.x * 256 + threadIdx.x;
  int c = (int)((i / NPIX) % CCH);
  float y = pre[i];
  out[i] = fmaxf(gam[c] * (y - mean[c]) * istd[c] + bet[c], 0.0f);
}

// ---------------------------------------------------------------------------
// Host orchestration
// ---------------------------------------------------------------------------
extern "C" void kernel_launch(void* const* d_in, const int* in_sizes, int n_in,
                              void* d_out, int out_size, void* d_ws,
                              size_t ws_size, hipStream_t stream) {
  const float* input_feature = (const float*)d_in[0];
  const float* fb = (const float*)d_in[1];
  const float* fc = (const float*)d_in[2];
  const float* fd = (const float*)d_in[3];
  const float* Wa = (const float*)d_in[4];
  const float* Wb = (const float*)d_in[5];
  const float* Wc = (const float*)d_in[6];
  const float* Wd = (const float*)d_in[7];
  const float* fuse_w = (const float*)d_in[8];
  const float* fuse_b = (const float*)d_in[9];
  const float* fuse_gamma = (const float*)d_in[10];
  const float* fuse_beta = (const float*)d_in[11];
  const float* gamma_cam = (const float*)d_in[12];
  const float* out_w = (const float*)d_in[13];
  const float* out_b = (const float*)d_in[14];
  const float* out_gamma = (const float*)d_in[15];
  const float* out_beta = (const float*)d_in[16];
  float* out = (float*)d_out;

  // Workspace layout
  const size_t SZ_QK  = (size_t)4 * BB * CCH * NPIX * sizeof(__bf16); // 113 MB
  const size_t SZ_CAT = SZ_QK;                                        // 113 MB
  const size_t SZ_PRE = (size_t)BB * CCH * NPIX * sizeof(float);      // 56.6 MB (reused)
  const size_t SZ_MID = (size_t)BB * CCH * NPIX * sizeof(__bf16);     // 28.3 MB
  const size_t SZ_ENE = (size_t)32 * CCH * CCH * sizeof(float);       // 294 KB

  char* ws = (char*)d_ws;
  __bf16* qkbuf = (__bf16*)(ws);
  __bf16* cat   = (__bf16*)(ws + SZ_QK);
  float*  pre   = (float*)(ws + SZ_QK + SZ_CAT);           // fuse_pre then out_pre
  __bf16* mid   = (__bf16*)(ws + SZ_QK + SZ_CAT + SZ_PRE);
  float*  energy = (float*)(ws + SZ_QK + SZ_CAT + SZ_PRE + SZ_MID);
  __bf16* attn  = (__bf16*)((char*)energy + SZ_ENE);
  float*  stats = (float*)((char*)attn + (size_t)32 * CCH * 64 * sizeof(__bf16));
  float* fuseSum = stats, * fuseSsq = stats + 48;
  float* outSum = stats + 96, * outSsq = stats + 144;
  float* fuseMean = stats + 192, * fuseIstd = stats + 240;
  float* outMean = stats + 288, * outIstd = stats + 336;

  // Zero accumulators (graph-capturable async memsets)
  hipMemsetAsync(energy, 0, SZ_ENE, stream);
  hipMemsetAsync(stats, 0, 192 * sizeof(float), stream);

  // 1) q/K projections
  qk_gemm_kernel<<<dim3(NPIX / 128, 32), 256, 0, stream>>>(
      input_feature, fb, fc, fd, Wa, Wb, Wc, Wd, qkbuf);
  // 2) energy
  energy_kernel<<<dim3(32, 32), 288, 0, stream>>>(qkbuf, energy);
  // 3) softmax
  softmax_kernel<<<32, 64, 0, stream>>>(energy, attn);
  // 4) out4 -> cat
  out4_gemm_kernel<<<dim3(NPIX / 128, 32), 256, 0, stream>>>(qkbuf, attn, cat);
  // 5) fuse 1x1 conv + BN stats  (pre = fuse_pre)
  fuse_conv_kernel<<<dim3(NPIX / 64, BB), 384, 0, stream>>>(
      cat, fuse_w, fuse_b, pre, fuseSum, fuseSsq);
  stats_kernel<<<1, 64, 0, stream>>>(fuseSum, fuseSsq, fuseMean, fuseIstd,
                                     1.0f / ((float)BB * NPIX));
  // 6) BN + relu + residual -> mid (bf16)
  {
    size_t total = (size_t)BB * CCH * NPIX;
    bn_mid_kernel<<<(unsigned)(total / 256), 256, 0, stream>>>(
        pre, input_feature, fuseMean, fuseIstd, fuse_gamma, fuse_beta,
        gamma_cam, mid);
  }
  // 7) 3x3 conv + BN stats  (pre reused as out_pre)
  out_conv_kernel<<<dim3(192 / 64, 192, BB), 384, 0, stream>>>(
      mid, out_w, out_b, pre, outSum, outSsq);
  stats_kernel<<<1, 64, 0, stream>>>(outSum, outSsq, outMean, outIstd,
                                     1.0f / ((float)BB * NPIX));
  // 8) final BN + relu -> d_out
  {
    size_t total = (size_t)BB * CCH * NPIX;
    bn_out_kernel<<<(unsigned)(total / 256), 256, 0, stream>>>(
        pre, outMean, outIstd, out_gamma, out_beta, out);
  }
}
